// HCL_71090298683968
// MI455X (gfx1250) — matrix operand order repaired and checked by reading, not compiled, not run on hardware
//
#include <hip/hip_runtime.h>
#include <math.h>

// ---------- types ----------
typedef __attribute__((ext_vector_type(16))) __bf16        v16bf;
typedef __attribute__((ext_vector_type(8)))  float         v8f;
typedef __attribute__((ext_vector_type(8)))  unsigned int  v8u;
typedef __attribute__((ext_vector_type(4)))  unsigned int  v4u;
typedef __attribute__((ext_vector_type(8)))  int           v8i;
typedef __attribute__((ext_vector_type(4)))  int           v4i;

// ---------- problem constants ----------
#define BATCH 8
#define HH 256
#define WW 256
#define HWN 65536          // H*W
#define LHW 16384          // (H/2)*(W/2)
#define C0 64
#define C1 128
#define CF 32
#define CIN 96
#define CHID 256
#define CP 128
#define NREP 3
#define NIT 10

// ---------- workspace byte offsets ----------
#define OFF_W1B   0ull                                   // 256*96*2   = 49152
#define OFF_W2B   49152ull                               // 128*256*2  = 65536
#define OFF_FUSED 114688ull                              // 8*16384*32*2 = 8388608
#define OFF_FPM   8503296ull                             // 8*65536*128*2 = 134217728
#define OFF_CENTS 142721024ull                           // 8*3*2*128*4 = 24576
#define OFF_GACC  142745600ull                           // 8*3*128*4   = 12288
#define OFF_TOTAL 142757888ull                           // 8*128*4     = 4096
#define OFF_SCORE 142761984ull                           // 8*3*4 -> pad 128
#define OFF_BEST  142762112ull                           // 8*4

// ---------- LDS layout for k_mlp (dynamic, 204800 bytes) ----------
#define SX_OFF   0        // [128][96]  bf16 = 24576
#define SH_OFF   24576    // [128][256] bf16 = 65536
#define SW1_OFF  90112    // [256][96]  bf16 = 49152
#define SW2_OFF  139264   // [128][256] bf16 = 65536
#define SMEM_MLP 204800
#define SP_OFF   0        // [128][128] f32 aliases sX/sH after barrier

// ---------- helpers ----------
__device__ __forceinline__ unsigned short f2bf(float f) {
  unsigned u = __builtin_bit_cast(unsigned, f);
  unsigned r = u + 0x7FFFu + ((u >> 16) & 1u);
  return (unsigned short)(r >> 16);
}
__device__ __forceinline__ float bf2f(unsigned short h) {
  return __builtin_bit_cast(float, ((unsigned)h) << 16);
}
__device__ __forceinline__ unsigned pack2bf(float a, float b) {
  return (unsigned)f2bf(a) | ((unsigned)f2bf(b) << 16);
}

// A-matrix 16x32 bf16 fragment (ISA 7.12.2): lane half selects K-offset 8,
// v<4 -> K = 2v, v>=4 -> K = 16+2(v-4)
__device__ __forceinline__ v16bf frag_a(const unsigned short* __restrict__ base,
                                        int row0, int stride, int k0, int lane) {
  int m = lane & 15, half = lane >> 4;
  v8u u;
#pragma unroll
  for (int v = 0; v < 8; ++v) {
    int k = k0 + ((v < 4) ? (2 * v) : (16 + 2 * (v - 4))) + 8 * half;
    u[v] = *(const unsigned*)(base + (size_t)(row0 + m) * stride + k);
  }
  return __builtin_bit_cast(v16bf, u);
}
// B-matrix 32x16 bf16 fragment: same K order as frag_a (half selects K-offset 8)
__device__ __forceinline__ v16bf frag_b(const unsigned short* __restrict__ base,
                                        int n0, int stride, int k0, int lane) {
  int n = lane & 15, half = lane >> 4;
  v8u u;
#pragma unroll
  for (int v = 0; v < 8; ++v) {
    int k = k0 + ((v < 4) ? (2 * v) : (16 + 2 * (v - 4))) + 8 * half;
    u[v] = *(const unsigned*)(base + (size_t)(n0 + n) * stride + k);
  }
  return __builtin_bit_cast(v16bf, u);
}

#define WMMA_BF16(a, b, c) \
  __builtin_amdgcn_wmma_f32_16x16x32_bf16(false, (a), false, (b), (short)0, (c), false, false)

// ---------- TDM: 1-D bf16 tile global -> LDS ----------
#if __has_builtin(__builtin_amdgcn_tensor_load_to_lds)
#define HAS_TDM 1
__device__ __forceinline__ void tdm_load_bf16(unsigned lds_off, const void* gptr,
                                              unsigned nelem) {
  unsigned long long ga = (unsigned long long)(size_t)gptr;
  v4u g0;
  g0[0] = 1u;                                        // count=1, user mode
  g0[1] = lds_off;                                   // lds_addr
  g0[2] = (unsigned)(ga & 0xFFFFFFFFu);              // global_addr[31:0]
  g0[3] = (unsigned)((ga >> 32) & 0x1FFFFFFu) | (2u << 30);  // addr[56:32] | type=2
  v8i g1;
  g1[0] = (int)(1u << 16);                           // data_size=1 (2 bytes)
  g1[1] = (int)((nelem & 0xFFFFu) << 16);            // tensor_dim0[15:0]
  g1[2] = (int)(((nelem >> 16) & 0xFFFFu) | (1u << 16)); // td0 hi | tensor_dim1=1
  g1[3] = (int)((nelem & 0xFFFFu) << 16);            // tile_dim0 = nelem
  g1[4] = 1;                                         // tile_dim1 = 1
  g1[5] = (int)nelem;                                // tensor_dim0_stride
  g1[6] = 0;
  g1[7] = 0;
  v4i z4 = {0, 0, 0, 0};
#if __clang_major__ >= 23
  v8i z8 = {0, 0, 0, 0, 0, 0, 0, 0};
  __builtin_amdgcn_tensor_load_to_lds(g0, g1, z4, z4, z8, 0);
#else
  __builtin_amdgcn_tensor_load_to_lds(g0, g1, z4, z4, 0);
#endif
}
#endif

// =====================================================================
// Kernel: convert w1 [256][96] and w2 [128][256] to bf16
// =====================================================================
__global__ __launch_bounds__(256) void k_cvt(const float* __restrict__ w1,
                                             const float* __restrict__ w2,
                                             unsigned short* __restrict__ w1b,
                                             unsigned short* __restrict__ w2b) {
  int i = blockIdx.x * 256 + threadIdx.x;
  if (i < CHID * CIN) w1b[i] = f2bf(w1[i]);
  if (i < CP * CHID)  w2b[i] = f2bf(w2[i]);
}

// =====================================================================
// Kernel: fuse conv 128->32 at low res, store bf16 pixel-major [B][LHW][32]
// =====================================================================
__global__ __launch_bounds__(256) void k_fuse(const float* __restrict__ feat1,
                                              const float* __restrict__ fw,
                                              const float* __restrict__ fb,
                                              unsigned short* __restrict__ fusedb) {
  __shared__ float sw[CF * C1];
  __shared__ float sb[CF];
  for (int i = threadIdx.x; i < CF * C1; i += 256) sw[i] = fw[i];
  if (threadIdx.x < CF) sb[threadIdx.x] = fb[threadIdx.x];
  __syncthreads();
  int lp = blockIdx.x * 256 + threadIdx.x;      // 8*16384 total
  int b = lp >> 14, i = lp & (LHW - 1);
  float acc[CF];
#pragma unroll
  for (int o = 0; o < CF; ++o) acc[o] = sb[o];
  for (int c = 0; c < C1; ++c) {
    float x = feat1[((size_t)b * C1 + c) * LHW + i];
#pragma unroll
    for (int o = 0; o < CF; ++o) acc[o] = fmaf(x, sw[o * C1 + c], acc[o]);
  }
  unsigned* dst = (unsigned*)(fusedb + (size_t)lp * CF);
#pragma unroll
  for (int o = 0; o < CF / 2; ++o) dst[o] = pack2bf(acc[2 * o], acc[2 * o + 1]);
}

// =====================================================================
// Fused MLP: pack x tile -> GEMM1(WMMA bf16) -> GELU -> GEMM2 -> L2 norm
// 128 pixels per workgroup; 8 waves, each owns a 16-row M-strip.
// =====================================================================
__global__ __launch_bounds__(256) void k_mlp(const float* __restrict__ feat0,
                                             const unsigned short* __restrict__ fusedb,
                                             const unsigned short* __restrict__ w1b,
                                             const unsigned short* __restrict__ w2b,
                                             const float* __restrict__ b1,
                                             const float* __restrict__ b2,
                                             float* __restrict__ proj,
                                             unsigned short* __restrict__ fpm) {
  extern __shared__ char smem[];
  unsigned short* sX  = (unsigned short*)(smem + SX_OFF);
  unsigned short* sH  = (unsigned short*)(smem + SH_OFF);
  unsigned short* sW1 = (unsigned short*)(smem + SW1_OFF);
  unsigned short* sW2 = (unsigned short*)(smem + SW2_OFF);
  float*          sP  = (float*)(smem + SP_OFF);

  const int tid = threadIdx.x;
  const int wid = tid >> 5, lane = tid & 31;
  const unsigned pix0 = blockIdx.x << 7;        // 128 pixels / block
  const int b = (int)(pix0 >> 16);
  const int hw0 = (int)(pix0 & (HWN - 1));

  // ---- weights into LDS via Tensor Data Mover (wave 0 only; EXEC ignored) ----
#ifdef HAS_TDM
  if (wid == 0) {
    tdm_load_bf16(SW1_OFF, w1b, CHID * CIN);
    tdm_load_bf16(SW2_OFF, w2b, CP * CHID);
    __builtin_amdgcn_s_wait_tensorcnt(0);
  }
#else
  for (int i = tid; i < CHID * CIN; i += 256) sW1[i] = w1b[i];
  for (int i = tid; i < CP * CHID; i += 256)  sW2[i] = w2b[i];
#endif

  // ---- build x tile [128][96] bf16: feat0 (64ch, strided) + fused (32ch) ----
  for (int i = tid; i < 128 * C0; i += 256) {
    int p = i & 127, c = i >> 7;
    float v = feat0[((size_t)b * C0 + c) * HWN + hw0 + p];
    sX[p * CIN + c] = f2bf(v);
  }
  for (int i = tid; i < 128 * CF; i += 256) {
    int p = i & 127, c = i >> 7;
    int hw = hw0 + p, hr = hw >> 8, wc = hw & 255;
    int lw = (hr >> 1) * (WW / 2) + (wc >> 1);
    sX[p * CIN + C0 + c] = fusedb[((size_t)b * LHW + lw) * CF + c];
  }
  __syncthreads();

  const int m0 = wid * 16;
  const int half = lane >> 4, ln = lane & 15;
  v8f zero = {0.f, 0.f, 0.f, 0.f, 0.f, 0.f, 0.f, 0.f};

  // ---- GEMM1: [128x96] @ [96x256] -> h, GELU, store bf16 to sH ----
  {
    v8f acc[16];
#pragma unroll
    for (int nt = 0; nt < 16; ++nt) acc[nt] = zero;
#pragma unroll
    for (int ks = 0; ks < 3; ++ks) {
      v16bf a = frag_a(sX, m0, CIN, ks * 32, lane);
#pragma unroll
      for (int nt = 0; nt < 16; ++nt) {
        v16bf bb = frag_b(sW1, nt * 16, CIN, ks * 32, lane);
        acc[nt] = WMMA_BF16(a, bb, acc[nt]);
      }
    }
#pragma unroll
    for (int nt = 0; nt < 16; ++nt) {
      int col = nt * 16 + ln;
      float bias = b1[col];
#pragma unroll
      for (int r = 0; r < 8; ++r) {
        int row = m0 + r + 8 * half;
        float v = acc[nt][r] + bias;
        v = 0.5f * v * (1.0f + erff(v * 0.70710678118654752f));   // exact GELU
        sH[row * CHID + col] = f2bf(v);
      }
    }
  }
  __syncthreads();

  // ---- GEMM2: [128x256] @ [256x128] -> proj, L2 norm per pixel row ----
  v8f acc2[8];
#pragma unroll
  for (int nt = 0; nt < 8; ++nt) acc2[nt] = zero;
#pragma unroll
  for (int ks = 0; ks < 8; ++ks) {
    v16bf a = frag_a(sH, m0, CHID, ks * 32, lane);
#pragma unroll
    for (int nt = 0; nt < 8; ++nt) {
      v16bf bb = frag_b(sW2, nt * 16, CHID, ks * 32, lane);
      acc2[nt] = WMMA_BF16(a, bb, acc2[nt]);
    }
  }
#pragma unroll
  for (int nt = 0; nt < 8; ++nt) {
    float bias = b2[nt * 16 + ln];
#pragma unroll
    for (int r = 0; r < 8; ++r) acc2[nt][r] += bias;
  }
  // per-row sum of squares: 16 lanes (same half) share a row -> xor-shuffle 1,2,4,8
  float inv[8];
#pragma unroll
  for (int r = 0; r < 8; ++r) {
    float s = 0.f;
#pragma unroll
    for (int nt = 0; nt < 8; ++nt) s += acc2[nt][r] * acc2[nt][r];
    s += __shfl_xor(s, 1, 32);
    s += __shfl_xor(s, 2, 32);
    s += __shfl_xor(s, 4, 32);
    s += __shfl_xor(s, 8, 32);
    inv[r] = 1.0f / fmaxf(sqrtf(s), 1e-12f);
  }
  __syncthreads();   // everyone done with sX/sH before sP aliases them

  // ---- normalized proj -> LDS [128 px][128 ch] f32 ----
#pragma unroll
  for (int nt = 0; nt < 8; ++nt)
#pragma unroll
    for (int r = 0; r < 8; ++r)
      sP[(size_t)(m0 + r + 8 * half) * CP + nt * 16 + ln] = acc2[nt][r] * inv[r];
  __syncthreads();

  // ---- write channel-major f32 proj to d_out ----
  {
    int c = tid >> 1, p0 = (tid & 1) * 64;
    float* dst = proj + ((size_t)b * CP + c) * HWN + hw0 + p0;
    for (int i = 0; i < 64; ++i) dst[i] = sP[(size_t)(p0 + i) * CP + c];
  }
  // ---- write pixel-major bf16 copy for k-means ----
  {
    int p = tid >> 1, h2 = (tid & 1) * 64;
    unsigned* dst = (unsigned*)(fpm + ((size_t)b * HWN + hw0 + p) * CP + h2);
    const float* src = sP + (size_t)p * CP + h2;
    for (int i = 0; i < 32; ++i) dst[i] = pack2bf(src[2 * i], src[2 * i + 1]);
  }
}

// =====================================================================
// K-means init: deterministic centroid picks, zero accumulators/scores
// =====================================================================
__global__ __launch_bounds__(256) void k_init(const unsigned short* __restrict__ fpm,
                                              float* __restrict__ cents,
                                              float* __restrict__ gacc,
                                              float* __restrict__ total,
                                              float* __restrict__ scores) {
  int t = threadIdx.x;
  for (int i = t; i < BATCH * NREP * 2 * CP; i += 256) {
    int d = i & 127, k = (i >> 7) & 1, br = i >> 8;
    int b = br / NREP, r = br % NREP;
    int idx = (r * 6151 + k * 997 + 13) & (HWN - 1);
    cents[i] = bf2f(fpm[((size_t)b * HWN + idx) * CP + d]);
  }
  for (int i = t; i < BATCH * NREP * CP; i += 256) gacc[i] = 0.f;
  for (int i = t; i < BATCH * CP; i += 256) total[i] = 0.f;
  if (t < BATCH * NREP) scores[t] = 0.f;
}

// =====================================================================
// total[b][d] = sum over points of f  (iteration-invariant; c0 = total - c1)
// =====================================================================
__global__ __launch_bounds__(256) void k_total(const unsigned short* __restrict__ fpm,
                                               float* __restrict__ total) {
  extern __shared__ char sm0[];
  unsigned short* ft = (unsigned short*)sm0;     // [256][128]
  int tid = threadIdx.x;
  int b = blockIdx.x >> 8, pt0 = (blockIdx.x & 255) << 8;
  const unsigned* src = (const unsigned*)(fpm + (((size_t)b << 16) + pt0 + tid) * CP);
  unsigned* drow = (unsigned*)(ft + (size_t)tid * CP);
#pragma unroll 8
  for (int j = 0; j < 64; ++j) drow[j] = src[j];
  __syncthreads();
  int d = tid & 127, ph = tid >> 7;
  float s = 0.f;
  for (int p = ph * 128; p < ph * 128 + 128; ++p) s += bf2f(ft[p * CP + d]);
  atomicAdd(total + b * CP + d, s);
}

// =====================================================================
// One k-means iteration, phase A: labels + per-dim masked column sums
// =====================================================================
__global__ __launch_bounds__(256) void k_assign(const unsigned short* __restrict__ fpm,
                                                const float* __restrict__ cents,
                                                float* __restrict__ gacc) {
  extern __shared__ char sm1[];
  unsigned short* ft = (unsigned short*)sm1;           // [256][128] = 64KB
  float* sc = (float*)(sm1 + 65536);                   // [6][128]
  unsigned char* lab = (unsigned char*)(sm1 + 65536 + 3072);
  int tid = threadIdx.x;
  int b = blockIdx.x >> 8, pt0 = (blockIdx.x & 255) << 8;
  for (int i = tid; i < 6 * CP; i += 256) sc[i] = cents[b * 6 * CP + i];
  __syncthreads();
  const unsigned* src = (const unsigned*)(fpm + (((size_t)b << 16) + pt0 + tid) * CP);
  unsigned* drow = (unsigned*)(ft + (size_t)tid * CP);
  float dot[6] = {0.f, 0.f, 0.f, 0.f, 0.f, 0.f};
  for (int j = 0; j < 64; ++j) {
    unsigned u = src[j];
    drow[j] = u;
    float f0 = bf2f((unsigned short)u), f1 = bf2f((unsigned short)(u >> 16));
    int d = 2 * j;
#pragma unroll
    for (int rk = 0; rk < 6; ++rk)
      dot[rk] += f0 * sc[rk * CP + d] + f1 * sc[rk * CP + d + 1];
  }
  unsigned lb = 0;
#pragma unroll
  for (int r = 0; r < NREP; ++r) lb |= (dot[2 * r + 1] > dot[2 * r] ? 1u : 0u) << r;
  lab[tid] = (unsigned char)lb;
  __syncthreads();
  int d = tid & 127, ph = tid >> 7;
  float a0 = 0.f, a1 = 0.f, a2 = 0.f;
  for (int p = ph * 128; p < ph * 128 + 128; ++p) {
    float fv = bf2f(ft[p * CP + d]);
    unsigned l = lab[p];
    if (l & 1u) a0 += fv;
    if (l & 2u) a1 += fv;
    if (l & 4u) a2 += fv;
  }
  atomicAdd(gacc + ((size_t)b * NREP + 0) * CP + d, a0);
  atomicAdd(gacc + ((size_t)b * NREP + 1) * CP + d, a1);
  atomicAdd(gacc + ((size_t)b * NREP + 2) * CP + d, a2);
}

// =====================================================================
// Phase B: cents[k=1] = norm(c1), cents[k=0] = norm(total - c1); re-zero gacc
// =====================================================================
__global__ __launch_bounds__(128) void k_update(float* __restrict__ cents,
                                                float* __restrict__ gacc,
                                                const float* __restrict__ total) {
  __shared__ float red[8];
  int br = blockIdx.x;           // b*3 + r  (24 blocks)
  int b = br / NREP;
  int d = threadIdx.x;
  int wid = d >> 5, lane = d & 31;
  float c1 = gacc[(size_t)br * CP + d];
  float c0 = total[(size_t)b * CP + d] - c1;
  gacc[(size_t)br * CP + d] = 0.f;

  float v = c1 * c1;
  for (int m = 1; m < 32; m <<= 1) v += __shfl_xor(v, m, 32);
  if (lane == 0) red[wid] = v;
  __syncthreads();
  float s1 = red[0] + red[1] + red[2] + red[3];
  __syncthreads();
  v = c0 * c0;
  for (int m = 1; m < 32; m <<= 1) v += __shfl_xor(v, m, 32);
  if (lane == 0) red[wid] = v;
  __syncthreads();
  float s0 = red[0] + red[1] + red[2] + red[3];

  cents[((size_t)br * 2 + 0) * CP + d] = c0 / fmaxf(sqrtf(s0), 1e-12f);
  cents[((size_t)br * 2 + 1) * CP + d] = c1 / fmaxf(sqrtf(s1), 1e-12f);
}

// =====================================================================
// Score per restart: sum of best-similarity (argmax over restarts == mean)
// =====================================================================
__global__ __launch_bounds__(256) void k_score(const unsigned short* __restrict__ fpm,
                                               const float* __restrict__ cents,
                                               float* __restrict__ scores) {
  __shared__ float sc[6 * CP];
  __shared__ float red[NREP][8];
  int tid = threadIdx.x;
  int b = blockIdx.x >> 8, pt0 = (blockIdx.x & 255) << 8;
  for (int i = tid; i < 6 * CP; i += 256) sc[i] = cents[b * 6 * CP + i];
  __syncthreads();
  const unsigned* src = (const unsigned*)(fpm + (((size_t)b << 16) + pt0 + tid) * CP);
  float dot[6] = {0.f, 0.f, 0.f, 0.f, 0.f, 0.f};
  for (int j = 0; j < 64; ++j) {
    unsigned u = src[j];
    float f0 = bf2f((unsigned short)u), f1 = bf2f((unsigned short)(u >> 16));
    int d = 2 * j;
#pragma unroll
    for (int rk = 0; rk < 6; ++rk)
      dot[rk] += f0 * sc[rk * CP + d] + f1 * sc[rk * CP + d + 1];
  }
  int wid = tid >> 5, lane = tid & 31;
#pragma unroll
  for (int r = 0; r < NREP; ++r) {
    float m = fmaxf(dot[2 * r], dot[2 * r + 1]);
    for (int msk = 1; msk < 32; msk <<= 1) m += __shfl_xor(m, msk, 32);
    if (lane == 0) red[r][wid] = m;
  }
  __syncthreads();
  if (tid < NREP) {
    float s = 0.f;
    for (int w = 0; w < 8; ++w) s += red[tid][w];
    atomicAdd(scores + b * NREP + tid, s);
  }
}

__global__ void k_best(const float* __restrict__ scores, int* __restrict__ best) {
  int t = threadIdx.x;
  if (t < BATCH) {
    const float* s = scores + t * NREP;
    int r = 0;
    if (s[1] > s[r]) r = 1;
    if (s[2] > s[r]) r = 2;
    best[t] = r;
  }
}

// =====================================================================
// Final: softmax(temp=0.1) over 2 clusters of best restart -> masks
// =====================================================================
__global__ __launch_bounds__(256) void k_final(const unsigned short* __restrict__ fpm,
                                               const float* __restrict__ cents,
                                               const int* __restrict__ best,
                                               float* __restrict__ outp,
                                               float* __restrict__ outc) {
  __shared__ float sc[2 * CP];
  int tid = threadIdx.x;
  int b = blockIdx.x >> 8, pt0 = (blockIdx.x & 255) << 8;
  int r = best[b];
  const float* cb = cents + ((size_t)(b * NREP + r) * 2) * CP;
  if (tid < 2 * CP) sc[tid] = cb[tid];
  __syncthreads();
  const unsigned* src = (const unsigned*)(fpm + (((size_t)b << 16) + pt0 + tid) * CP);
  float d0 = 0.f, d1 = 0.f;
  for (int j = 0; j < 64; ++j) {
    unsigned u = src[j];
    float f0 = bf2f((unsigned short)u), f1 = bf2f((unsigned short)(u >> 16));
    int d = 2 * j;
    d0 += f0 * sc[d] + f1 * sc[d + 1];
    d1 += f0 * sc[CP + d] + f1 * sc[CP + d + 1];
  }
  float l0 = d0 * 10.f, l1 = d1 * 10.f;          // / TEMP
  int lbl = (l1 > l0) ? 1 : 0;
  float conf = 1.f / (1.f + __expf(-fabsf(l1 - l0)));
  size_t idx = ((size_t)b << 16) + pt0 + tid;
  outp[idx] = (float)lbl;
  outc[idx] = (conf >= 0.95f) ? 1.f : 0.f;
}

// =====================================================================
extern "C" void kernel_launch(void* const* d_in, const int* in_sizes, int n_in,
                              void* d_out, int out_size, void* d_ws, size_t ws_size,
                              hipStream_t stream) {
  (void)in_sizes; (void)n_in; (void)out_size; (void)ws_size;
  const float* feat0  = (const float*)d_in[0];
  const float* feat1  = (const float*)d_in[1];
  const float* fuse_w = (const float*)d_in[2];
  const float* fuse_b = (const float*)d_in[3];
  const float* w1     = (const float*)d_in[4];
  const float* b1     = (const float*)d_in[5];
  const float* w2     = (const float*)d_in[6];
  const float* b2     = (const float*)d_in[7];

  char* ws = (char*)d_ws;
  unsigned short* w1b    = (unsigned short*)(ws + OFF_W1B);
  unsigned short* w2b    = (unsigned short*)(ws + OFF_W2B);
  unsigned short* fusedb = (unsigned short*)(ws + OFF_FUSED);
  unsigned short* fpm    = (unsigned short*)(ws + OFF_FPM);
  float* cents  = (float*)(ws + OFF_CENTS);
  float* gacc   = (float*)(ws + OFF_GACC);
  float* total  = (float*)(ws + OFF_TOTAL);
  float* scores = (float*)(ws + OFF_SCORE);
  int*   best   = (int*)(ws + OFF_BEST);

  float* proj  = (float*)d_out;
  float* predm = proj + (size_t)BATCH * CP * HWN;     // 67108864
  float* confm = predm + (size_t)BATCH * HWN;         // +524288

  k_cvt <<<dim3(128),  dim3(256), 0, stream>>>(w1, w2, w1b, w2b);
  k_fuse<<<dim3(512),  dim3(256), 0, stream>>>(feat1, fuse_w, fuse_b, fusedb);
  k_mlp <<<dim3(4096), dim3(256), SMEM_MLP, stream>>>(feat0, fusedb, w1b, w2b,
                                                      b1, b2, proj, fpm);
  k_init <<<dim3(1),    dim3(256), 0, stream>>>(fpm, cents, gacc, total, scores);
  k_total<<<dim3(2048), dim3(256), 65536, stream>>>(fpm, total);
  for (int it = 0; it < NIT; ++it) {
    k_assign<<<dim3(2048), dim3(256), 65536 + 3072 + 256, stream>>>(fpm, cents, gacc);
    k_update<<<dim3(24),   dim3(128), 0, stream>>>(cents, gacc, total);
  }
  k_score<<<dim3(2048), dim3(256), 0, stream>>>(fpm, cents, scores);
  k_best <<<dim3(1),    dim3(64),  0, stream>>>(scores, best);
  k_final<<<dim3(2048), dim3(256), 0, stream>>>(fpm, cents, best, predm, confm);
}
